// MultiHeadAttention_29076928594706
// MI455X (gfx1250) — matrix operand order repaired and checked by reading, not compiled, not run on hardware
//
#include <hip/hip_runtime.h>

// ---- problem constants ----
#define D_MODEL 1024
#define NUM_HEADS 16
#define DK 64
#define BB 2
#define SS 2048
#define MM (BB * SS) // 4096 token rows

typedef __attribute__((ext_vector_type(16))) __bf16 v16bf;
typedef __attribute__((ext_vector_type(8)))  __bf16 v8bf;
typedef __attribute__((ext_vector_type(8)))  float  v8f;

// A-operand (16x32 bf16) per-lane load: lane group g (lane>>4) holds
// K {g*16 .. g*16+15} contiguous; p already includes g*16.
__device__ __forceinline__ v16bf load_a16x32(const __bf16* p) {
  v8bf lo = *(const v8bf*)(p);
  v8bf hi = *(const v8bf*)(p + 8);
  return __builtin_shufflevector(lo, hi, 0, 1, 2, 3, 4, 5, 6, 7,
                                 8, 9, 10, 11, 12, 13, 14, 15);
}

__device__ __forceinline__ v8f wmma_bf16(v16bf a, v16bf b, v8f c) {
  return __builtin_amdgcn_wmma_f32_16x16x32_bf16(
      /*neg_a=*/false, a, /*neg_b=*/false, b,
      /*c_mod=*/(short)0, c, /*reuse_a=*/false, /*reuse_b=*/false);
}

// ---------------- fp32 -> bf16 conversion ----------------
__global__ void cvt_f32_bf16(const float* __restrict__ src,
                             __bf16* __restrict__ dst, int n) {
  int i = blockIdx.x * blockDim.x + threadIdx.x;
  if (i < n) dst[i] = (__bf16)src[i];
}

// ------- GEMM: C[m,e] = sum_d A[m,d] * W[e,d] + bias[e] ---------------------
// One wave per 32x64 output tile (B fragments reused across two A fragments).
// mode: 0 -> bf16 [b,h,s,64]   1 -> bf16 [b,h,64,s]   2 -> fp32 [m, D_MODEL]
__global__ __launch_bounds__(32)
void gemm_xwT(const __bf16* __restrict__ A, const __bf16* __restrict__ W,
              const float* __restrict__ bias,
              __bf16* __restrict__ outb, float* __restrict__ outf, int mode) {
  const int m0 = blockIdx.x * 32;
  const int n0 = blockIdx.y * 64;
  const int lane = threadIdx.x;
  const int half = lane >> 4;
  const int lr = lane & 15;

  v8f c[2][4];
#pragma unroll
  for (int mi = 0; mi < 2; ++mi)
#pragma unroll
    for (int t = 0; t < 4; ++t) c[mi][t] = v8f{};

  const __bf16* arow0 = A + (size_t)(m0 + lr) * D_MODEL;
  const __bf16* arow1 = arow0 + (size_t)16 * D_MODEL;

  for (int k0 = 0; k0 < D_MODEL; k0 += 32) {
    v16bf a0 = load_a16x32(arow0 + k0 + half * 16);
    v16bf a1 = load_a16x32(arow1 + k0 + half * 16);
    const __bf16* wbase = W + (size_t)(n0 + lr) * D_MODEL + k0 + half * 16;
#pragma unroll
    for (int t = 0; t < 4; ++t) {
      v16bf b = *(const v16bf*)(wbase + (size_t)t * 16 * D_MODEL);
      c[0][t] = wmma_bf16(a0, b, c[0][t]);
      c[1][t] = wmma_bf16(a1, b, c[1][t]);
    }
  }

#pragma unroll
  for (int t = 0; t < 4; ++t) {
    const int e = n0 + t * 16 + lr;
    const float bv = bias[e];
#pragma unroll
    for (int mi = 0; mi < 2; ++mi) {
#pragma unroll
      for (int r = 0; r < 8; ++r) {
        const int m = m0 + mi * 16 + r + 8 * half;
        const float v = c[mi][t][r] + bv;
        if (mode == 2) {
          outf[(size_t)m * D_MODEL + e] = v;
        } else {
          const int b = m >> 11, s = m & (SS - 1);
          const int h = e >> 6, d = e & 63;
          const size_t idx =
              (mode == 0)
                  ? (((size_t)(b * NUM_HEADS + h) * SS + s) * DK + d)
                  : (((size_t)(b * NUM_HEADS + h) * DK + d) * SS + s);
          outb[idx] = (__bf16)v;
        }
      }
    }
  }
}

// ---------------- flash attention per (b, h, 16-row q tile) -----------------
// 64 key-columns per online-softmax round: 8 score WMMAs + 8 ctx WMMAs per
// round amortize the shfl-reduction/exp VALU work.
__global__ __launch_bounds__(32)
void attn_kernel(const __bf16* __restrict__ Qb, const __bf16* __restrict__ Kb,
                 const __bf16* __restrict__ Vtb, const int* __restrict__ mask,
                 __bf16* __restrict__ Ctx) {
  const int q0 = blockIdx.x * 16;
  const int h = blockIdx.y;
  const int b = blockIdx.z;
  const int bh = b * NUM_HEADS + h;
  const __bf16* Q = Qb + (size_t)bh * SS * DK;
  const __bf16* K = Kb + (size_t)bh * SS * DK;
  const __bf16* Vt = Vtb + (size_t)bh * DK * SS;
  const int* Mk = mask + (size_t)b * SS * SS;

  const int lane = threadIdx.x;
  const int half = lane >> 4;
  const int lr = lane & 15;

  // Preload Q A-fragments (16x64 split into two 16x32 chunks).
  const __bf16* qrow = Q + (size_t)(q0 + lr) * DK;
  const v16bf qa0 = load_a16x32(qrow + 0 + half * 16);
  const v16bf qa1 = load_a16x32(qrow + 32 + half * 16);

  float mrow[8], lrow[8];
#pragma unroll
  for (int r = 0; r < 8; ++r) { mrow[r] = -1e30f; lrow[r] = 0.0f; }
  v8f acc[4] = {v8f{}, v8f{}, v8f{}, v8f{}};

  __shared__ __bf16 Plds[16 * 64]; // 2 KB staging tile: P rows -> A fragment

  for (int k0 = 0; k0 < SS; k0 += 64) {
    if (k0 + 64 < SS) { // prefetch next K/V chunk
      __builtin_prefetch((const void*)(K + (size_t)(k0 + 64 + lane * 2) * DK), 0, 3);
      __builtin_prefetch((const void*)(Vt + (size_t)(lane << 1) * SS + k0 + 64), 0, 3);
    }

    // ---- scores: four 16x16 tiles, K-dim = 64 via two WMMAs each ----
    v8f s[4];
#pragma unroll
    for (int t = 0; t < 4; ++t) {
      const __bf16* krow = K + (size_t)(k0 + t * 16 + lr) * DK;
      v16bf kb0 = *(const v16bf*)(krow + 0 + half * 16);
      v16bf kb1 = *(const v16bf*)(krow + 32 + half * 16);
      v8f cz = {};
      cz = wmma_bf16(qa0, kb0, cz);
      cz = wmma_bf16(qa1, kb1, cz);
      s[t] = cz;
    }

    // ---- scale + mask ----
    const float sc = 0.125f; // 1/sqrt(64)
#pragma unroll
    for (int t = 0; t < 4; ++t) {
#pragma unroll
      for (int r = 0; r < 8; ++r) {
        const int qi = q0 + r + 8 * half;
        const int ki = k0 + t * 16 + lr;
        const int mv = Mk[(size_t)qi * SS + ki];
        const float v = s[t][r] * sc;
        s[t][r] = (mv == 0) ? -1e9f : v;
      }
    }

    // ---- online softmax (row lives in a 16-lane half, VGPR index r) ----
#pragma unroll
    for (int r = 0; r < 8; ++r) {
      float v = fmaxf(fmaxf(s[0][r], s[1][r]), fmaxf(s[2][r], s[3][r]));
#pragma unroll
      for (int off = 1; off <= 8; off <<= 1)
        v = fmaxf(v, __shfl_xor(v, off, 32));
      const float mn = fmaxf(mrow[r], v);
      const float corr = __expf(mrow[r] - mn);
      mrow[r] = mn;
      float rs = 0.0f;
#pragma unroll
      for (int t = 0; t < 4; ++t) {
        const float p = __expf(s[t][r] - mn);
        s[t][r] = p;
        rs += p;
      }
#pragma unroll
      for (int off = 1; off <= 8; off <<= 1)
        rs += __shfl_xor(rs, off, 32);
      lrow[r] = lrow[r] * corr + rs;
#pragma unroll
      for (int t = 0; t < 4; ++t) acc[t][r] *= corr;
    }

    // ---- re-lay P (C layout) -> A fragments via LDS ----
    __syncthreads();
#pragma unroll
    for (int t = 0; t < 4; ++t)
#pragma unroll
      for (int r = 0; r < 8; ++r)
        Plds[(r + 8 * half) * 64 + t * 16 + lr] = (__bf16)s[t][r];
    __syncthreads();
    const v16bf pa0 = load_a16x32(&Plds[lr * 64 + half * 16]);
    const v16bf pa1 = load_a16x32(&Plds[lr * 64 + 32 + half * 16]);

    // ---- ctx += P(16x64) x V[k0:k0+64, 0:64] (Vt rows contiguous) ----
#pragma unroll
    for (int t = 0; t < 4; ++t) {
      const __bf16* vrow = Vt + (size_t)(t * 16 + lr) * SS + k0 + half * 16;
      v16bf vb0 = *(const v16bf*)(vrow);
      v16bf vb1 = *(const v16bf*)(vrow + 32);
      acc[t] = wmma_bf16(pa0, vb0, acc[t]);
      acc[t] = wmma_bf16(pa1, vb1, acc[t]);
    }
  }

  // ---- normalize + store ctx bf16 [m, D_MODEL] ----
#pragma unroll
  for (int r = 0; r < 8; ++r) {
    const float inv = (lrow[r] > 0.0f) ? (1.0f / lrow[r]) : 0.0f;
    const int row = q0 + r + 8 * half;
#pragma unroll
    for (int t = 0; t < 4; ++t) {
      const int d = t * 16 + lr;
      Ctx[(size_t)(b * SS + row) * D_MODEL + h * DK + d] =
          (__bf16)(acc[t][r] * inv);
    }
  }
}

// ---------------- host-side launcher ----------------
extern "C" void kernel_launch(void* const* d_in, const int* in_sizes, int n_in,
                              void* d_out, int out_size, void* d_ws,
                              size_t ws_size, hipStream_t stream) {
  (void)in_sizes; (void)n_in; (void)out_size; (void)ws_size;
  const float* hs = (const float*)d_in[0];
  const int* mask = (const int*)d_in[1];
  const float* Wq = (const float*)d_in[2];
  const float* bq = (const float*)d_in[3];
  const float* Wk = (const float*)d_in[4];
  const float* bk = (const float*)d_in[5];
  const float* Wv = (const float*)d_in[6];
  const float* bv = (const float*)d_in[7];
  const float* Wo = (const float*)d_in[8];
  const float* bo = (const float*)d_in[9];
  float* out = (float*)d_out;

  char* ws = (char*)d_ws;
  const size_t MB = 1ull << 20;
  __bf16* Xb  = (__bf16*)(ws + 0 * MB);   // 8 MB
  __bf16* Wqb = (__bf16*)(ws + 8 * MB);   // 2 MB
  __bf16* Wkb = (__bf16*)(ws + 10 * MB);  // 2 MB
  __bf16* Wvb = (__bf16*)(ws + 12 * MB);  // 2 MB
  __bf16* Wob = (__bf16*)(ws + 14 * MB);  // 2 MB
  __bf16* Qb  = (__bf16*)(ws + 16 * MB);  // 8 MB  [b,h,s,64]
  __bf16* Kb  = (__bf16*)(ws + 24 * MB);  // 8 MB  [b,h,s,64]
  __bf16* Vtb = (__bf16*)(ws + 32 * MB);  // 8 MB  [b,h,64,s]
  __bf16* Ctx = (__bf16*)(ws + 40 * MB);  // 8 MB  [m, 1024]

  const int nX = MM * D_MODEL;      // 4,194,304
  const int nW = D_MODEL * D_MODEL; // 1,048,576
  cvt_f32_bf16<<<nX / 256, 256, 0, stream>>>(hs, Xb, nX);
  cvt_f32_bf16<<<nW / 256, 256, 0, stream>>>(Wq, Wqb, nW);
  cvt_f32_bf16<<<nW / 256, 256, 0, stream>>>(Wk, Wkb, nW);
  cvt_f32_bf16<<<nW / 256, 256, 0, stream>>>(Wv, Wvb, nW);
  cvt_f32_bf16<<<nW / 256, 256, 0, stream>>>(Wo, Wob, nW);

  dim3 g(MM / 32, D_MODEL / 64); // 128 x 16 tiles of 32x64
  gemm_xwT<<<g, 32, 0, stream>>>(Xb, Wqb, bq, Qb, nullptr, 0);
  gemm_xwT<<<g, 32, 0, stream>>>(Xb, Wkb, bk, Kb, nullptr, 0);
  gemm_xwT<<<g, 32, 0, stream>>>(Xb, Wvb, bv, Vtb, nullptr, 1);

  attn_kernel<<<dim3(SS / 16, NUM_HEADS, BB), 32, 0, stream>>>(Qb, Kb, Vtb,
                                                               mask, Ctx);

  gemm_xwT<<<g, 32, 0, stream>>>(Ctx, Wob, bo, nullptr, out, 2);
}
